// RHMM_6597069766978
// MI455X (gfx1250) — matrix operand order repaired and checked by reading, not compiled, not run on hardware
//
#include <hip/hip_runtime.h>
#include <hip/hip_bf16.h>
#include <stdint.h>

// ---------- CDNA5 types ----------
typedef __attribute__((ext_vector_type(16))) _Float16 v16h;
typedef __attribute__((ext_vector_type(8)))  float    v8f;
typedef __attribute__((ext_vector_type(4)))  unsigned int v4u;
typedef __attribute__((ext_vector_type(8)))  int      v8i;
typedef __attribute__((ext_vector_type(4)))  int      v4i;

#define HMM_B 8
#define HMM_T 256
#define HMM_N 128
#define HMM_K 64
#define TILE_STRIDE 129   // 128 + 1 DW pad -> conflict-free LDS rows & cols

#if defined(__HIP_DEVICE_COMPILE__) && __has_builtin(__builtin_amdgcn_tensor_load_to_lds) && __has_builtin(__builtin_amdgcn_s_wait_tensorcnt)
#define USE_TDM 1
#else
#define USE_TDM 0
#endif

// =====================================================================
// Kernel 1: emission logits = obs[B*T,K] @ W[K,N] + b  via v_wmma_f32_16x16x32_f16
// blockDim = 128 (4 waves), each wave computes a 16-row x 128-col stripe.
// =====================================================================
__global__ void emit_gemm_wmma(const float* __restrict__ obs,
                               const float* __restrict__ W,
                               const float* __restrict__ bvec,
                               float* __restrict__ logits) {
  const int lane = threadIdx.x & 31;
  const int wave = threadIdx.x >> 5;
  const int ln   = lane & 15;
  const bool hi  = lane >= 16;
  const int m0   = (blockIdx.x * 4 + wave) * 16;   // row tile base in [0, 2048)

  v8f acc[8];
#pragma unroll
  for (int jt = 0; jt < 8; ++jt)
#pragma unroll
    for (int r = 0; r < 8; ++r) acc[jt][r] = 0.0f;

#pragma unroll
  for (int ks = 0; ks < 2; ++ks) {                 // K = 64 -> two 16x16x32 steps
    const int k0 = ks * 32;
    // A fragment: 16x32 f16. lane(lo): M=lane, K in {0..15}; lane(hi): M=lane-16, K in {16..31}
    v16h a;
    const float* arow = obs + (size_t)(m0 + ln) * HMM_K + k0;
#pragma unroll
    for (int v = 0; v < 8; ++v) {
      const int kb = (hi ? 16 : 0) + 2 * v;
      a[2 * v]     = (_Float16)arow[kb];
      a[2 * v + 1] = (_Float16)arow[kb + 1];
    }
#pragma unroll
    for (int jt = 0; jt < 8; ++jt) {
      const int n0 = jt * 16;
      // B fragment: 32x16 f16. VGPR v holds K = (hi?16:0)+2v+{0,1}, N = lane%16
      v16h bf;
      const int kb = k0 + (hi ? 16 : 0);
#pragma unroll
      for (int v = 0; v < 8; ++v) {
        bf[2 * v]     = (_Float16)W[(size_t)(kb + 2 * v) * HMM_N + n0 + ln];
        bf[2 * v + 1] = (_Float16)W[(size_t)(kb + 2 * v + 1) * HMM_N + n0 + ln];
      }
      acc[jt] = __builtin_amdgcn_wmma_f32_16x16x32_f16(
          /*neg_a=*/false, a, /*neg_b=*/false, bf,
          /*c_mod=*/(short)0, acc[jt], /*reuse_a=*/false, /*reuse_b=*/false);
    }
  }
  // D layout: VGPR r -> M = r + 8*hi, N = lane%16
#pragma unroll
  for (int jt = 0; jt < 8; ++jt) {
    const int n0 = jt * 16;
#pragma unroll
    for (int r = 0; r < 8; ++r) {
      const int row = m0 + r + (hi ? 8 : 0);
      const int col = n0 + ln;
      logits[(size_t)row * HMM_N + col] = acc[jt][r] + bvec[col];
    }
  }
}

// =====================================================================
// Kernel 2: in-place row-wise log-softmax over N=128 (one block per row)
// =====================================================================
__global__ void logsoftmax_rows(float* __restrict__ le) {
  __shared__ float red[HMM_N];
  const int row = blockIdx.x;
  const int j = threadIdx.x;
  const float x = le[(size_t)row * HMM_N + j];
  red[j] = x; __syncthreads();
#pragma unroll
  for (int s = 64; s > 0; s >>= 1) { if (j < s) red[j] = fmaxf(red[j], red[j + s]); __syncthreads(); }
  const float m = red[0]; __syncthreads();
  red[j] = __expf(x - m); __syncthreads();
#pragma unroll
  for (int s = 64; s > 0; s >>= 1) { if (j < s) red[j] += red[j + s]; __syncthreads(); }
  const float ls = __logf(red[0]);
  le[(size_t)row * HMM_N + j] = x - m - ls;
}

// =====================================================================
// TDM tile load: 128x128 f32 tile (contiguous 64KB) -> LDS with 1-DW pad
// every 128 DW (row stride 129). Descriptor per CDNA5 ISA.
// This toolchain exposes the 6-arg builtin:
//   (uint32x4 g0, int32x8 g1, int32x4 g2, int32x4 g3, int32x8 aux, i32 cpol)
// =====================================================================
__device__ __forceinline__ void tdm_load_tile(const float* gsrc, unsigned int lds_off) {
#if USE_TDM
  const uint64_t ga = (uint64_t)(uintptr_t)gsrc;
  v4u g0;
  g0[0] = 1u;                                            // count=1, user mode
  g0[1] = lds_off;                                       // lds_addr (bytes)
  g0[2] = (unsigned int)(ga & 0xFFFFFFFFu);              // global_addr[31:0]
  g0[3] = (unsigned int)((ga >> 32) & 0x1FFFFFFu)        // global_addr[56:32]
        | (2u << 30);                                    // type=2 ("image")
  v8i g1;
  g1[0] = (2 << 16)                                      // data_size = 4B
        | (1 << 20)                                      // pad_enable
        | (6 << 22);                                     // pad_interval: 128 DW
                                                         // pad_amount bits[31:25]=0 -> 1 DW
  g1[1] = (HMM_N << 16);                                 // tensor_dim0 = 128 (lo16)
  g1[2] = (HMM_N << 16);                                 // tensor_dim1 = 128 (lo16)
  g1[3] = (HMM_N << 16);                                 // tile_dim0 = 128
  g1[4] = HMM_N;                                         // tile_dim1 = 128, tile_dim2 = 0
  g1[5] = HMM_N;                                         // tensor_dim0_stride = 128 (lo32)
  g1[6] = 0;
  g1[7] = 0;
  v4i g2 = {0, 0, 0, 0};
  v4i g3 = {0, 0, 0, 0};
  v8i aux = {0, 0, 0, 0, 0, 0, 0, 0};
  __builtin_amdgcn_tensor_load_to_lds(g0, g1, g2, g3, aux, 0);
#else
  // Fallback: wave-0 synchronous padded copy (32 lanes)
  extern __shared__ float smem_fb[];
  float* dst = (float*)((char*)smem_fb + (lds_off - (unsigned int)(uintptr_t)smem_fb));
  const int lane = threadIdx.x & 31;
  for (int idx = lane; idx < HMM_N * HMM_N; idx += 32) {
    const int r = idx >> 7, c = idx & 127;
    dst[r * TILE_STRIDE + c] = gsrc[idx];
  }
#endif
}

__device__ __forceinline__ void tdm_wait(int n) {
#if USE_TDM
  if (n == 0) __builtin_amdgcn_s_wait_tensorcnt(0);
  else        __builtin_amdgcn_s_wait_tensorcnt(1);
#else
  (void)n;
#endif
}

// =====================================================================
// Kernel 3: scaled linear-domain HMM forward recursion.
// One block (128 threads, 4 waves) per batch b. Double-buffered TDM tiles.
//   la[j] <- le[t,j] + m + log( sum_k q_k * trans[t,b,k,j] ),
//   q_k = exp(la_k - m) / rowsum_k   (folds the L1 normalization)
// =====================================================================
__global__ void hmm_forward(const float* __restrict__ le,     // [B,T,N] log emissions
                            const float* __restrict__ trans,  // [T,B,N,N]
                            const float* __restrict__ start,  // [B,N]
                            const int* __restrict__ dur,      // [B]
                            float* __restrict__ out) {        // [B]
  extern __shared__ float smem[];
  float* tile0 = smem;                          // 128*129
  float* tile1 = smem + HMM_N * TILE_STRIDE;    // 128*129
  float* la    = smem + 2 * HMM_N * TILE_STRIDE;
  float* q     = la + HMM_N;
  float* red   = q + HMM_N;

  const int b = blockIdx.x;
  const int j = threadIdx.x;                    // 0..127
  const int d = dur[b];                         // in [1, T]

  // t = 0: la = log_emit[:,0,:] + log(start_prob)
  la[j] = le[((size_t)b * HMM_T) * HMM_N + j] + __logf(start[(size_t)b * HMM_N + j]);
  __syncthreads();

  const unsigned int off0 = (unsigned int)(uintptr_t)tile0;
  const unsigned int off1 = (unsigned int)(uintptr_t)tile1;
  const bool issuer = (threadIdx.x < 32);       // wave 0 owns the TDM queue

  if (d > 1 && issuer)
    tdm_load_tile(trans + ((size_t)1 * HMM_B + b) * (HMM_N * HMM_N), off0);

  for (int t = 1; t < d; ++t) {
    float* cur = (t & 1) ? tile0 : tile1;       // tile t: t=1 -> tile0, t=2 -> tile1, ...
    if (issuer) {
      if (t + 1 < d) {
        tdm_load_tile(trans + ((size_t)(t + 1) * HMM_B + b) * (HMM_N * HMM_N),
                      (t & 1) ? off1 : off0);
        tdm_wait(1);                            // in-order: older load (tile t) complete
      } else {
        tdm_wait(0);
      }
    }
    __syncthreads();                            // tile t visible to all waves

    // m = max_k la_k
    red[j] = la[j]; __syncthreads();
#pragma unroll
    for (int s = 64; s > 0; s >>= 1) { if (j < s) red[j] = fmaxf(red[j], red[j + s]); __syncthreads(); }
    const float m = red[0]; __syncthreads();

    // rowsum for row k=j (LDS stride 129 -> conflict-free), then q_k
    {
      const float* rowp = cur + j * TILE_STRIDE;
      float rs = 0.0f;
#pragma unroll 8
      for (int k2 = 0; k2 < HMM_N; ++k2) rs += rowp[k2];
      q[j] = __expf(la[j] - m) / rs;
    }
    __syncthreads();

    // matvec: acc_j = sum_k q_k * tile[k][j]  (column read also conflict-free)
    float acc = 0.0f;
#pragma unroll 8
    for (int k = 0; k < HMM_N; ++k) acc += q[k] * cur[k * TILE_STRIDE + j];
    const float nv = le[((size_t)b * HMM_T + t) * HMM_N + j] + m + __logf(acc);
    __syncthreads();
    la[j] = nv;
    __syncthreads();
  }

  // output: logsumexp_j la[j] at t = d-1
  red[j] = la[j]; __syncthreads();
#pragma unroll
  for (int s = 64; s > 0; s >>= 1) { if (j < s) red[j] = fmaxf(red[j], red[j + s]); __syncthreads(); }
  const float m2 = red[0]; __syncthreads();
  red[j] = __expf(la[j] - m2); __syncthreads();
#pragma unroll
  for (int s = 64; s > 0; s >>= 1) { if (j < s) red[j] += red[j + s]; __syncthreads(); }
  if (j == 0) out[b] = m2 + __logf(red[0]);
}

// =====================================================================
extern "C" void kernel_launch(void* const* d_in, const int* in_sizes, int n_in,
                              void* d_out, int out_size, void* d_ws, size_t ws_size,
                              hipStream_t stream) {
  (void)in_sizes; (void)n_in; (void)out_size; (void)ws_size;
  const float* obs   = (const float*)d_in[0];   // [B,T,K]
  const int*   dur   = (const int*)d_in[1];     // [B]
  const float* trans = (const float*)d_in[2];   // [T,B,N,N]
  const float* start = (const float*)d_in[3];   // [B,N]
  const float* W     = (const float*)d_in[4];   // [K,N]
  const float* bvec  = (const float*)d_in[5];   // [N]
  float* out = (float*)d_out;                   // [B,1]
  float* le  = (float*)d_ws;                    // [B*T, N] = 1 MiB scratch

  // 1) emission logits via WMMA: 2048 rows -> 128 stripes, 4 waves/block
  emit_gemm_wmma<<<32, 128, 0, stream>>>(obs, W, bvec, le);
  // 2) log-softmax per row
  logsoftmax_rows<<<HMM_B * HMM_T, HMM_N, 0, stream>>>(le);
  // 3) forward recursion, double-buffered TDM transition tiles
  const size_t shmem = (2 * HMM_N * TILE_STRIDE + 3 * HMM_N) * sizeof(float);
  hmm_forward<<<HMM_B, HMM_N, shmem, stream>>>(le, trans, start, dur, out);
}
